// CrossBlock_36060545417850
// MI455X (gfx1250) — matrix operand order repaired and checked by reading, not compiled, not run on hardware
//
#include <hip/hip_runtime.h>

// ---------------------------------------------------------------------------
// Types
// ---------------------------------------------------------------------------
typedef __bf16 bf16;
typedef bf16  v16bf __attribute__((ext_vector_type(16)));
typedef float v8f   __attribute__((ext_vector_type(8)));
typedef float v4f   __attribute__((ext_vector_type(4)));
typedef unsigned int v4u __attribute__((ext_vector_type(4)));
typedef unsigned short u16;

union U32B { v4u q[2]; v16bf v; u16 e[16]; };   // 32 bytes: one WMMA A/B operand
union U16B { v4u q;    u16 e[8];  };            // 16 bytes

__device__ __forceinline__ u16 f2u16(float f) {
    unsigned u = __builtin_bit_cast(unsigned, f);
    u += 0x7fffu + ((u >> 16) & 1u);            // round-to-nearest-even bf16
    return (u16)(u >> 16);
}
__device__ __forceinline__ float u162f(u16 s) {
    unsigned u = ((unsigned)s) << 16;
    return __builtin_bit_cast(float, u);
}
__device__ __forceinline__ v8f wmma_bf16(v16bf a, v16bf b, v8f c) {
    // D(16x16,f32) = A(16x32,bf16) x B(32x16,bf16) + C
    return __builtin_amdgcn_wmma_f32_16x16x32_bf16(false, a, false, b, (short)0, c, false, false);
}
__device__ __forceinline__ v8f zero_v8f() {
    v8f z;
#pragma unroll
    for (int i = 0; i < 8; ++i) z[i] = 0.0f;
    return z;
}

// LDS byte offset of a generic pointer into shared memory: the LDS aperture
// keeps the workgroup-relative byte address in ADDR[31:0] (ISA 10.2), so a
// 32-bit truncation of the flat address is the VDST operand for async ops.
__device__ __forceinline__ unsigned lds_off(const void* p) {
    return (unsigned)(unsigned long long)p;
}
// ASYNCcnt-tracked 16B global -> LDS copy (per-lane addresses on both sides).
__device__ __forceinline__ void async_b128(const void* g, unsigned l) {
    asm volatile("global_load_async_to_lds_b128 %0, %1, off"
                 :: "v"(l), "v"((unsigned long long)g)
                 : "memory");
}

// ---------------------------------------------------------------------------
// Problem constants
// ---------------------------------------------------------------------------
#define BN   8
#define LN_  2048
#define CN   256
#define HN   4
#define DN   64
#define MROWS (BN * LN_)        // 16384 rows per side

// ---------------------------------------------------------------------------
// Weight transpose + convert:  src f32 (K x N) -> dst bf16 (N x K), row offset
// ---------------------------------------------------------------------------
__global__ __launch_bounds__(256) void wprep_kernel(const float* __restrict__ src,
                                                    u16* __restrict__ dst,
                                                    int K, int N, int rowoff, int ldd) {
    int idx = blockIdx.x * 256 + threadIdx.x;
    if (idx >= K * N) return;
    int k = idx / N, n = idx % N;
    dst[(size_t)(n + rowoff) * ldd + k] = f2u16(src[idx]);
}

// ---------------------------------------------------------------------------
// f32 -> bf16 pack (4 elements / thread)
// ---------------------------------------------------------------------------
__global__ __launch_bounds__(256) void pack_kernel(const float* __restrict__ in,
                                                   u16* __restrict__ out) {
    size_t i = ((size_t)blockIdx.x * 256 + threadIdx.x) * 4;
    v4f f = *(const v4f*)(in + i);
    out[i + 0] = f2u16(f[0]);
    out[i + 1] = f2u16(f[1]);
    out[i + 2] = f2u16(f[2]);
    out[i + 3] = f2u16(f[3]);
}

// ---------------------------------------------------------------------------
// Generic bf16 WMMA GEMM: C(MxN,f32) = A(MxK) @ Wt(NxK)^T + bias
//   Block tile 128x128, 8 waves, wave tile 64x32 (4x2 WMMA tiles).
//   K tiles staged with global_load_async_to_lds_b128, double buffered:
//   tile k+1 streams in (ASYNCcnt) while tile k feeds the WMMAs.
//   Dual-A: k < KA1 reads A1 (lda=KA1), else A2 (lda=K-KA1)  (FFN concat trick)
//   MODE 0: head-scatter epilogue -> qk (scaled) / v tensors  (N=512)
//   MODE 1: bf16 row-major out (+bias)
//   MODE 2: f32 out = acc + bias + residual  (final output)
// ---------------------------------------------------------------------------
template <int MODE>
__global__ __launch_bounds__(256) void gemm_kernel(
    const u16* __restrict__ A1, const u16* __restrict__ A2, int KA1,
    const u16* __restrict__ Wt, int K, int N,
    const float* __restrict__ bias1, const float* __restrict__ bias2,
    float qkScale,
    u16* __restrict__ outA, u16* __restrict__ outB,
    float* __restrict__ outF, const float* __restrict__ residual) {

    __shared__ u16 As[2][128 * 40];   // padded stride 40 elems = 80B, 16B-aligned rows
    __shared__ u16 Bs[2][128 * 40];

    const int tid  = threadIdx.x;
    const int lane = tid & 31, w = tid >> 5;
    const int wr = w & 1, wc = w >> 1;
    const int row0 = blockIdx.x * 128;
    const int col0 = blockIdx.y * 128;
    const int m15 = lane & 15, hi = lane >> 4;
    const int kb8 = hi * 8, kb16 = hi * 16;
    const int lrow = tid >> 1, lseg = tid & 1;

    v8f acc[4][2];
#pragma unroll
    for (int i = 0; i < 4; ++i)
#pragma unroll
        for (int j = 0; j < 2; ++j) acc[i][j] = zero_v8f();

    // issue one k-tile (128x32 of A and of Wt) as 4 async b128 per thread
    auto stage = [&](int kt, int buf) {
        const int k0 = kt * 32;
        const u16* src; int kk, lda;
        if (k0 < KA1) { src = A1; kk = k0;       lda = KA1;     }
        else          { src = A2; kk = k0 - KA1; lda = K - KA1; }
        const u16* ga = src + (size_t)(row0 + lrow) * lda + kk + lseg * 16;
        unsigned la = lds_off(&As[buf][lrow * 40 + lseg * 16]);
        async_b128(ga,     la);
        async_b128(ga + 8, la + 16);
        const u16* gb = Wt + (size_t)(col0 + lrow) * K + k0 + lseg * 16;
        unsigned lb = lds_off(&Bs[buf][lrow * 40 + lseg * 16]);
        async_b128(gb,     lb);
        async_b128(gb + 8, lb + 16);
        asm volatile("s_wait_xcnt 0x0" ::: "memory");   // addr regs safe to recycle
    };

    const int nk = K / 32;
    stage(0, 0);
    for (int kt = 0; kt < nk; ++kt) {
        __syncthreads();                       // buf[(kt+1)&1] no longer being read
        if (kt + 1 < nk) {
            stage(kt + 1, (kt + 1) & 1);
            asm volatile("s_wait_asynccnt 0x4" ::: "memory");  // tile kt drained
        } else {
            asm volatile("s_wait_asynccnt 0x0" ::: "memory");
        }
        __syncthreads();                       // tile kt visible to all waves

        const u16* Ab = As[kt & 1];
        const u16* Bb = Bs[kt & 1];
        v16bf aop[4], bop[2];
#pragma unroll
        for (int rt = 0; rt < 4; ++rt) {
            const v4u* p = (const v4u*)&Ab[(wr * 64 + rt * 16 + m15) * 40 + kb16];
            U32B u; u.q[0] = p[0]; u.q[1] = p[1];   // K run kb16..kb16+15
            aop[rt] = u.v;
        }
#pragma unroll
        for (int ct = 0; ct < 2; ++ct) {
            const v4u* p = (const v4u*)&Bb[(wc * 32 + ct * 16 + m15) * 40 + kb16];
            U32B u; u.q[0] = p[0]; u.q[1] = p[1];   // K run kb16..kb16+15
            bop[ct] = u.v;
        }
#pragma unroll
        for (int rt = 0; rt < 4; ++rt)
#pragma unroll
            for (int ct = 0; ct < 2; ++ct)
                acc[rt][ct] = wmma_bf16(aop[rt], bop[ct], acc[rt][ct]);
    }

    // epilogue — C layout: M = vg + 8*hi, N = lane&15
#pragma unroll
    for (int rt = 0; rt < 4; ++rt) {
        const int growb = row0 + wr * 64 + rt * 16;
#pragma unroll
        for (int ct = 0; ct < 2; ++ct) {
            const int col = col0 + wc * 32 + ct * 16 + m15;
#pragma unroll
            for (int vg = 0; vg < 8; ++vg) {
                const int row = growb + vg + 8 * hi;
                float val = acc[rt][ct][vg];
                if (MODE == 0) {
                    const int b_ = row >> 11;          // row / 2048
                    const int l  = row & 2047;
                    if (col < CN) {                    // qk half, scaled by D^-0.25
                        val = (val + bias1[col]) * qkScale;
                        const int h = col >> 6, d = col & 63;
                        outA[(((size_t)(b_ * HN + h)) * LN_ + l) * DN + d] = f2u16(val);
                    } else {                           // v half
                        const int c2 = col - CN;
                        val += bias2[c2];
                        const int h = c2 >> 6, d = c2 & 63;
                        outB[(((size_t)(b_ * HN + h)) * LN_ + l) * DN + d] = f2u16(val);
                    }
                } else if (MODE == 1) {
                    outA[(size_t)row * N + col] = f2u16(val + bias1[col]);
                } else {
                    const size_t idx = (size_t)row * N + col;
                    outF[idx] = val + bias1[col] + residual[idx];
                }
            }
        }
    }
}

// ---------------------------------------------------------------------------
// Bidirectional attention. One block = 32 query rows of one (dir, b, h).
// Full 32x2048 sim strip lives in LDS (~265KB of the WGP's 320KB).
// The exp pass repacks P as bf16 in place, so pass 2's A operands are two
// ds_load_b128 instead of 16 b32 loads + converts.
// dir 0: Q=qk0, K=qk1, V=v1 -> m0 ; dir 1: Q=qk1, K=qk0, V=v0 -> m1
// ---------------------------------------------------------------------------
#define SSTR 2068   // padded f32 stride for 2048-wide strip

__global__ __launch_bounds__(256) void attn_kernel(const u16* __restrict__ qk,
                                                   const u16* __restrict__ vv,
                                                   u16* __restrict__ mout) {
    extern __shared__ char smem[];
    float* strip  = (float*)smem;                               // 32 * SSTR f32
    u16*   VT     = (u16*)(smem + 32 * SSTR * 4);               // 64 x 64 (stride 72)
    float* rowmax = (float*)(smem + 32 * SSTR * 4 + 64 * 72 * 2);
    float* rowsum = rowmax + 32;
    float* red    = rowsum + 32;                                // 32 x 8

    const int tid = threadIdx.x, lane = tid & 31, w = tid >> 5;
    const int dir = blockIdx.z, bh = blockIdx.y;
    const size_t headSz = (size_t)LN_ * DN;
    const u16* Q  = qk + (size_t)(dir * (BN * HN) + bh) * headSz;
    const u16* Kp = qk + (size_t)((1 - dir) * (BN * HN) + bh) * headSz;
    const u16* Vp = vv + (size_t)((1 - dir) * (BN * HN) + bh) * headSz;
    const int qrow0 = blockIdx.x * 32;
    const int m15 = lane & 15, hi = lane >> 4, kb8 = hi * 8, kb16 = hi * 16;
    const int rt = w & 1;

    // Preload Q A-operands for this wave's 16 rows (K=64 -> 2 k-steps), kept in regs.
    U32B ua0, ua1;
    {
        const int qr = qrow0 + rt * 16 + m15;
        const v4u* p0 = (const v4u*)(Q + (size_t)qr * DN + kb16);
        ua0.q[0] = p0[0]; ua0.q[1] = p0[1];
        const v4u* p1 = (const v4u*)(Q + (size_t)qr * DN + 32 + kb16);
        ua1.q[0] = p1[0]; ua1.q[1] = p1[1];
    }

    // ---- Pass 1: sim strip = Q K^T (32 x 2048) ----
    {
        const int cw = w >> 1;                        // 4 column sectors of 512
        for (int ct = 0; ct < 32; ++ct) {
            const int c0 = cw * 512 + ct * 16;
            const int kr = c0 + m15;                  // key row = output column
            U32B ub0, ub1;
            const v4u* p0 = (const v4u*)(Kp + (size_t)kr * DN + kb16);
            ub0.q[0] = p0[0]; ub0.q[1] = p0[1];
            const v4u* p1 = (const v4u*)(Kp + (size_t)kr * DN + 32 + kb16);
            ub1.q[0] = p1[0]; ub1.q[1] = p1[1];
            v8f s = zero_v8f();
            s = wmma_bf16(ua0.v, ub0.v, s);
            s = wmma_bf16(ua1.v, ub1.v, s);
#pragma unroll
            for (int vg = 0; vg < 8; ++vg)
                strip[(rt * 16 + vg + 8 * hi) * SSTR + c0 + m15] = s[vg];
        }
    }
    __syncthreads();

    // ---- Softmax: row max, then exp in place repacked as bf16 ----
    // Packed layout: P[row][c] lives at u16 index row*2*SSTR + (c>>8)*512 + (c&255)
    {
        const int row = tid >> 3, seg = tid & 7;      // 8 threads per row
        float* rp = strip + row * SSTR + seg * 256;
        u16* rp16 = (u16*)rp;
        float mx = -3.0e38f;
        for (int c = 0; c < 256; ++c) mx = fmaxf(mx, rp[c]);
        red[row * 8 + seg] = mx;
        __syncthreads();
        if (seg == 0) {
            float m2 = red[row * 8];
            for (int j = 1; j < 8; ++j) m2 = fmaxf(m2, red[row * 8 + j]);
            rowmax[row] = m2;
        }
        __syncthreads();
        const float rm = rowmax[row];
        float s = 0.f;
        for (int cc = 0; cc < 256; cc += 32) {
            float tmp[32];
#pragma unroll
            for (int j = 0; j < 32; ++j) {
                float p = __expf(rp[cc + j] - rm);
                s += p; tmp[j] = p;
            }
            asm volatile("" ::: "memory");   // f32 reads before aliased u16 writes
#pragma unroll
            for (int j = 0; j < 32; ++j) rp16[cc + j] = f2u16(tmp[j]);
            asm volatile("" ::: "memory");
        }
        red[row * 8 + seg] = s;
        __syncthreads();
        if (seg == 0) {
            float t = 0.f;
            for (int j = 0; j < 8; ++j) t += red[row * 8 + j];
            rowsum[row] = t;
        }
    }

    // ---- Pass 2: O = P V (32 x 64), 64-key V chunks staged transposed ----
    const u16* pstr = (const u16*)strip;
    const int oc0 = (w >> 1) * 16;                    // this wave's 16 output cols
    v8f oacc = zero_v8f();
    for (int kk = 0; kk < LN_; kk += 64) {
        __syncthreads();
        {   // stage V[kk..kk+64][0..64] transposed -> VT[d][kloc], stride 72
            const int vr = tid >> 2;                  // key row 0..63
            const int cs = tid & 3;                   // 16-col segment
            const v4u* gv = (const v4u*)(Vp + (size_t)(kk + vr) * DN + cs * 16);
            U32B u; u.q[0] = gv[0]; u.q[1] = gv[1];
#pragma unroll
            for (int j = 0; j < 16; ++j) VT[(cs * 16 + j) * 72 + vr] = u.e[j];
        }
        __syncthreads();
#pragma unroll
        for (int s2 = 0; s2 < 2; ++s2) {
            const int c0 = kk + s2 * 32;
            // A operand: packed bf16 P, two b128 loads
            const u16* pa = pstr + (size_t)(rt * 16 + m15) * (2 * SSTR)
                                 + ((c0 >> 8) << 9) + (c0 & 255) + kb16;
            const v4u* ap = (const v4u*)pa;
            U32B ua; ua.q[0] = ap[0]; ua.q[1] = ap[1];
            // B operand: contiguous 16 bf16 from transposed V
            const v4u* bp = (const v4u*)&VT[(oc0 + m15) * 72 + s2 * 32 + kb16];
            U32B ub; ub.q[0] = bp[0]; ub.q[1] = bp[1];
            oacc = wmma_bf16(ua.v, ub.v, oacc);
        }
    }

    // ---- Epilogue: normalize by row sum, scatter to merged (B,L,C) bf16 ----
    {
        const int b_ = bh >> 2, h = bh & 3;
        const int d = oc0 + m15;
#pragma unroll
        for (int vg = 0; vg < 8; ++vg) {
            const int srow = rt * 16 + vg + 8 * hi;
            const int l = qrow0 + srow;
            const float val = oacc[vg] / rowsum[srow];
            mout[((size_t)dir * MROWS + (size_t)b_ * LN_ + l) * CN + h * DN + d] = f2u16(val);
        }
    }
}

// ---------------------------------------------------------------------------
// LayerNorm(512) + exact GELU, in place on bf16 hidden. One wave per row.
// ---------------------------------------------------------------------------
__global__ __launch_bounds__(256) void ln_gelu_kernel(u16* __restrict__ h,
                                                      const float* __restrict__ g,
                                                      const float* __restrict__ bta) {
    const int row  = blockIdx.x * 8 + (threadIdx.x >> 5);
    const int lane = threadIdx.x & 31;
    u16* hp = h + (size_t)row * 512;
    float x[16]; float s = 0.f, ss = 0.f;
#pragma unroll
    for (int i = 0; i < 16; ++i) {
        float f = u162f(hp[lane + i * 32]);
        x[i] = f; s += f; ss += f * f;
    }
#pragma unroll
    for (int off = 16; off > 0; off >>= 1) {
        s  += __shfl_xor(s,  off, 32);
        ss += __shfl_xor(ss, off, 32);
    }
    const float mu  = s * (1.0f / 512.0f);
    const float var = ss * (1.0f / 512.0f) - mu * mu;
    const float inv = rsqrtf(var + 1e-5f);
#pragma unroll
    for (int i = 0; i < 16; ++i) {
        const int c = lane + i * 32;
        float y = (x[i] - mu) * inv * g[c] + bta[c];
        y = 0.5f * y * (1.0f + erff(y * 0.70710678118654752f));
        hp[c] = f2u16(y);
    }
}

// ---------------------------------------------------------------------------
// Workspace layout (bytes)
// ---------------------------------------------------------------------------
constexpr size_t OFF_WT_QKV = 0;                                // bf16 [512][256]
constexpr size_t OFF_WT_OUT = OFF_WT_QKV + 512 * 256 * 2;       // bf16 [256][256]
constexpr size_t OFF_WT_1   = OFF_WT_OUT + 256 * 256 * 2;       // bf16 [512][512]
constexpr size_t OFF_WT_2   = OFF_WT_1   + 512 * 512 * 2;       // bf16 [256][512]
constexpr size_t OFF_XB     = OFF_WT_2   + 256 * 512 * 2;       // bf16 [2][16384][256]
constexpr size_t OFF_QK     = OFF_XB + (size_t)2 * MROWS * CN * 2;
constexpr size_t OFF_V      = OFF_QK + (size_t)2 * MROWS * CN * 2;
constexpr size_t OFF_M      = OFF_V  + (size_t)2 * MROWS * CN * 2;
constexpr size_t OFF_MP     = OFF_M  + (size_t)2 * MROWS * CN * 2;
constexpr size_t OFF_HB     = OFF_MP + (size_t)2 * MROWS * CN * 2;  // bf16 [2][16384][512]

extern "C" void kernel_launch(void* const* d_in, const int* in_sizes, int n_in,
                              void* d_out, int out_size, void* d_ws, size_t ws_size,
                              hipStream_t stream) {
    (void)in_sizes; (void)n_in; (void)out_size; (void)ws_size;

    const float* x0   = (const float*)d_in[0];
    const float* x1   = (const float*)d_in[1];
    const float* Wqk  = (const float*)d_in[2];
    const float* bqk  = (const float*)d_in[3];
    const float* Wv   = (const float*)d_in[4];
    const float* bv   = (const float*)d_in[5];
    const float* Wout = (const float*)d_in[6];
    const float* bout = (const float*)d_in[7];
    const float* W1   = (const float*)d_in[8];
    const float* b1   = (const float*)d_in[9];
    const float* lng  = (const float*)d_in[10];
    const float* lnb  = (const float*)d_in[11];
    const float* W2   = (const float*)d_in[12];
    const float* b2   = (const float*)d_in[13];
    float* out = (float*)d_out;

    char* ws = (char*)d_ws;
    u16* Wt_qkv = (u16*)(ws + OFF_WT_QKV);
    u16* Wt_out = (u16*)(ws + OFF_WT_OUT);
    u16* Wt_1   = (u16*)(ws + OFF_WT_1);
    u16* Wt_2   = (u16*)(ws + OFF_WT_2);
    u16* xb     = (u16*)(ws + OFF_XB);
    u16* qkbuf  = (u16*)(ws + OFF_QK);
    u16* vbuf   = (u16*)(ws + OFF_V);
    u16* mbuf   = (u16*)(ws + OFF_M);
    u16* mpbuf  = (u16*)(ws + OFF_MP);
    u16* hbuf   = (u16*)(ws + OFF_HB);

    const dim3 blk(256);
    const float scale = 0.35355339059327373f;   // 64^(-1/4)
    const size_t sideC = (size_t)MROWS * CN;    // elems per side, C-wide
    const size_t sideH = (size_t)MROWS * 512;   // elems per side, 2C-wide
    const size_t sideHead = (size_t)(BN * HN) * LN_ * DN;

    // 1) weights -> transposed bf16
    wprep_kernel<<<256,  blk, 0, stream>>>(Wqk,  Wt_qkv, 256, 256,   0, 256);
    wprep_kernel<<<256,  blk, 0, stream>>>(Wv,   Wt_qkv, 256, 256, 256, 256);
    wprep_kernel<<<256,  blk, 0, stream>>>(Wout, Wt_out, 256, 256,   0, 256);
    wprep_kernel<<<1024, blk, 0, stream>>>(W1,   Wt_1,   512, 512,   0, 512);
    wprep_kernel<<<512,  blk, 0, stream>>>(W2,   Wt_2,   512, 256,   0, 512);

    // 2) x -> bf16
    pack_kernel<<<4096, blk, 0, stream>>>(x0, xb);
    pack_kernel<<<4096, blk, 0, stream>>>(x1, xb + sideC);

    // 3) fused qk/v projection (N=512), scaled qk, head-scattered bf16 outputs
    for (int s = 0; s < 2; ++s) {
        const u16* a = xb + s * sideC;
        gemm_kernel<0><<<dim3(128, 4), blk, 0, stream>>>(
            a, a, 256, Wt_qkv, 256, 512, bqk, bv, scale,
            qkbuf + s * sideHead, vbuf + s * sideHead, nullptr, nullptr);
    }

    // 4) bidirectional attention (shared sim, dual softmax) -> merged m bf16
    const int attnLds = 32 * SSTR * 4 + 64 * 72 * 2 + (32 + 32 + 256) * 4; // 275200 B
    attn_kernel<<<dim3(LN_ / 32, BN * HN, 2), blk, attnLds, stream>>>(qkbuf, vbuf, mbuf);

    // 5) out projection
    for (int s = 0; s < 2; ++s) {
        const u16* a = mbuf + s * sideC;
        gemm_kernel<1><<<dim3(128, 2), blk, 0, stream>>>(
            a, a, 256, Wt_out, 256, 256, bout, nullptr, 1.0f,
            mpbuf + s * sideC, nullptr, nullptr, nullptr);
    }

    // 6) FFN part 1: h = [x | m] @ W1 + b1   (dual-A, no materialized concat)
    for (int s = 0; s < 2; ++s) {
        gemm_kernel<1><<<dim3(128, 4), blk, 0, stream>>>(
            xb + s * sideC, mpbuf + s * sideC, 256, Wt_1, 512, 512, b1, nullptr, 1.0f,
            hbuf + s * sideH, nullptr, nullptr, nullptr);
    }

    // 7) LayerNorm + exact GELU in place (both sides: 32768 rows)
    ln_gelu_kernel<<<4096, blk, 0, stream>>>(hbuf, lng, lnb);

    // 8) FFN part 2 + bias + residual -> f32 outputs
    for (int s = 0; s < 2; ++s) {
        const u16* a = hbuf + s * sideH;
        gemm_kernel<2><<<dim3(128, 2), blk, 0, stream>>>(
            a, a, 512, Wt_2, 512, 256, b2, nullptr, 1.0f,
            nullptr, nullptr, out + s * sideC, (s == 0 ? x0 : x1));
    }
}
